// Reshape_4329327035136
// MI455X (gfx1250) — matrix operand order repaired and checked by reading, not compiled, not run on hardware
//
#include <hip/hip_runtime.h>

typedef _Float16 f16;
typedef __attribute__((ext_vector_type(16))) _Float16 v16h;
typedef __attribute__((ext_vector_type(8)))  float    v8f;
typedef __attribute__((ext_vector_type(4)))  unsigned v4u;
typedef __attribute__((ext_vector_type(8)))  int      v8i_;
typedef __attribute__((ext_vector_type(4)))  int      v4i_;

#define B_      4
#define C_      256
#define HW_     9216
#define N_      4608
#define FOUT_   32
#define QK_TILES 288      // N/16 tiles per batch (16 pix x 32 feat each, 1KB)
#define V_TILES  2304     // (N/32)*(C/16) tiles per batch (32 key x 16 chan, 1KB)
#define KBLOCKS  144      // N/32

// workspace byte offsets (f16 payload)
#define Q_OFF   ((size_t)0)
#define K_OFF   ((size_t)B_*QK_TILES*1024)
#define V_OFF   ((size_t)2*B_*QK_TILES*1024)
#define WQF_OFF (V_OFF + (size_t)B_*V_TILES*1024)
#define WKF_OFF (WQF_OFF + (size_t)16*1024)
#define WVF_OFF (WKF_OFF + (size_t)16*1024)
// total ws need = WVF_OFF + 128KB  (~12 MB)

// ---------------------------------------------------------------------------
// TDM: 1-D contiguous global->LDS copy via Tensor Data Mover.
// D# per CDNA5 ISA ch.8: group0 {count=1, lds_addr, global_addr[56:0], type=2},
// group1 {data_size=8B, tensor_dim0=elems, tensor_dim1=1, tile_dim0=elems,
//         tile_dim1=1, tensor_dim0_stride=elems}. bytes must be multiple of 8,
// elems <= 65535.
// ---------------------------------------------------------------------------
__device__ __forceinline__ void tdm_load_1d(const void* gaddr, unsigned lds_off,
                                            unsigned bytes)
{
    unsigned long long ga = (unsigned long long)(uintptr_t)gaddr;
    unsigned elems = bytes >> 3;
    v4u g0;
    g0[0] = 1u;                                                  // count=1
    g0[1] = lds_off;                                             // lds_addr
    g0[2] = (unsigned)ga;                                        // global lo32
    g0[3] = (unsigned)((ga >> 32) & 0x01FFFFFFu) | 0x80000000u;  // hi25 | type=2
    v8i_ g1;
    g1[0] = (int)(3u << 16);                     // workgroup_mask=0, data_size=8B
    g1[1] = (int)((elems & 0xFFFFu) << 16);      // tensor_dim0 lo16
    g1[2] = (int)((elems >> 16) | (1u << 16));   // tensor_dim0 hi16, tensor_dim1=1
    g1[3] = (int)((elems & 0xFFFFu) << 16);      // tile_dim0
    g1[4] = 1;                                   // tile_dim1=1
    g1[5] = (int)elems;                          // tensor_dim0_stride lo32
    g1[6] = 0;
    g1[7] = 0;
    v4i_ z4 = {0, 0, 0, 0};
#if __clang_major__ >= 23
    v8i_ z8 = {0, 0, 0, 0, 0, 0, 0, 0};
    __builtin_amdgcn_tensor_load_to_lds(g0, g1, z4, z4, z8, 0);
#else
    __builtin_amdgcn_tensor_load_to_lds(g0, g1, z4, z4, 0);
#endif
}

// ---------------------------------------------------------------------------
// Elementwise: out0 = f ; out1 = f*(1+(1-mask)*gamma). Attention kernel later
// overwrites the idx_out columns of both halves.
// ---------------------------------------------------------------------------
__global__ __launch_bounds__(256)
void elw_kernel(const float* __restrict__ f, const float* __restrict__ mask,
                const float* __restrict__ gamma, float* __restrict__ out)
{
    const size_t total = (size_t)B_ * C_ * HW_;
    size_t i = ((size_t)blockIdx.x * 256 + threadIdx.x) * 4;
    if (i >= total) return;
    float g = gamma[0];
    float4 fv = *(const float4*)(f + i);
    size_t b = i / ((size_t)C_ * HW_);
    size_t p = i % (size_t)HW_;
    float4 mv = *(const float4*)(mask + b * HW_ + p);
    float4 o1;
    o1.x = fv.x * (1.0f + (1.0f - mv.x) * g);
    o1.y = fv.y * (1.0f + (1.0f - mv.y) * g);
    o1.z = fv.z * (1.0f + (1.0f - mv.z) * g);
    o1.w = fv.w * (1.0f + (1.0f - mv.w) * g);
    *(float4*)(out + i) = fv;
    *(float4*)(out + total + i) = o1;
}

// ---------------------------------------------------------------------------
// Weight prep: pack W^T (C-chunk x 16-out) f32 row-major into f16 WMMA
// B-fragment tiles. One wave per tile; tile id = cbi*nOutTiles + ot.
// ---------------------------------------------------------------------------
__global__ __launch_bounds__(128)
void wfrag_kernel(const float* __restrict__ W, f16* __restrict__ dst, int nOutTiles)
{
    int gw = (int)(blockIdx.x * 4 + (threadIdx.x >> 5));
    if (gw >= 8 * nOutTiles) return;
    int cbi  = gw / nOutTiles;
    int ot   = gw % nOutTiles;
    int lane = threadIdx.x & 31;
    int half = lane >> 4;
    int l16  = lane & 15;
    union { v16h h; f16 e[16]; } o;
    const float* wrow = &W[(size_t)(ot * 16 + l16) * C_ + cbi * 32 + 16 * half];
    #pragma unroll
    for (int v = 0; v < 8; ++v) {
        o.e[2 * v]     = (f16)wrow[2 * v];
        o.e[2 * v + 1] = (f16)wrow[2 * v + 1];
    }
    *(v16h*)(dst + (size_t)gw * 512 + lane * 16) = o.h;
}

// ---------------------------------------------------------------------------
// Q/K projection: out[n, o] = sum_c W[o,c]*f[c, idx[n]] + bias[o]
// 64 gathered pixels per block (4 waves x 16 pixels), K-loop over C in 32-chunks.
// layoutB=0 -> A-fragment tiles (Q), layoutB=1 -> B-fragment tiles (K^T).
// ---------------------------------------------------------------------------
__global__ __launch_bounds__(128)
void proj32_kernel(const float* __restrict__ f, const int* __restrict__ idx,
                   const f16* __restrict__ WF, const float* __restrict__ bias,
                   f16* __restrict__ outT, int layoutB)
{
    __shared__ f16 lds[64 * 264];   // gathered features [pix][chan], stride 264
    __shared__ f16 ldsO[64 * 40];   // projection outputs [pix][out], stride 40

    const int tid   = threadIdx.x;
    const int b     = blockIdx.x / 72;
    const int pbase = (blockIdx.x % 72) * 64;

    const int pix   = tid & 63;
    const int myidx = idx[pbase + pix];
    const float* fb = f + (size_t)b * C_ * HW_;
    #pragma unroll 4
    for (int it = 0; it < 128; ++it) {
        int chan = it * 2 + (tid >> 6);
        lds[pix * 264 + chan] = (f16)fb[(size_t)chan * HW_ + myidx];
    }
    __syncthreads();

    const int wave = tid >> 5;
    const int lane = tid & 31;
    const int half = lane >> 4;
    const int l16  = lane & 15;
    const int prow = wave * 16;

    v8f d0 = {}; v8f d1 = {};
    for (int cbi = 0; cbi < 8; ++cbi) {
        union { v16h h; unsigned u[8]; } a;
        const f16* arow = &lds[(prow + l16) * 264 + cbi * 32];
        #pragma unroll
        for (int v = 0; v < 8; ++v) {
            int kv = 16 * half + 2 * v;
            a.u[v] = *(const unsigned*)(arow + kv);
        }
        v16h b0 = *(const v16h*)(WF + (size_t)(cbi * 2)     * 512 + lane * 16);
        v16h b1 = *(const v16h*)(WF + (size_t)(cbi * 2 + 1) * 512 + lane * 16);
        d0 = __builtin_amdgcn_wmma_f32_16x16x32_f16(false, a.h, false, b0, (short)0, d0, false, false);
        d1 = __builtin_amdgcn_wmma_f32_16x16x32_f16(false, a.h, false, b1, (short)0, d1, false, false);
    }
    float bb0 = bias[l16], bb1 = bias[16 + l16];
    #pragma unroll
    for (int r = 0; r < 8; ++r) {
        int m = prow + 8 * half + r;
        ldsO[m * 40 + l16]      = (f16)(d0[r] + bb0);
        ldsO[m * 40 + 16 + l16] = (f16)(d1[r] + bb1);
    }
    asm volatile("s_wait_dscnt 0x0" ::: "memory");

    union { v16h h; unsigned u[8]; } o;
    const f16* orow = &ldsO[(prow + l16) * 40];
    #pragma unroll
    for (int v = 0; v < 8; ++v) {
        int col = 16 * half + 2 * v;
        o.u[v] = *(const unsigned*)(orow + col);
    }
    int tt = (pbase >> 4) + wave;
    *(v16h*)(outT + (size_t)(b * QK_TILES + tt) * 512 + lane * 16) = o.h;
}

// ---------------------------------------------------------------------------
// V projection: v[d, n] = sum_c Wv[d,c]*f[c, idx_in[n]] + bv[d]
// Stored as V^T in B-fragment tiles (32 keys x 16 chans, 1KB each).
// ---------------------------------------------------------------------------
__global__ __launch_bounds__(128)
void projv_kernel(const float* __restrict__ f, const int* __restrict__ idx,
                  const f16* __restrict__ WVF, const float* __restrict__ bv,
                  f16* __restrict__ outV)
{
    __shared__ f16 lds[256 * 72];   // phase1: [pix][chan] stride 264; phase2: [chan][key] stride 72

    const int tid   = threadIdx.x;
    const int b     = blockIdx.x / 72;
    const int pbase = (blockIdx.x % 72) * 64;

    const int pix   = tid & 63;
    const int myidx = idx[pbase + pix];
    const float* fb = f + (size_t)b * C_ * HW_;
    #pragma unroll 4
    for (int it = 0; it < 128; ++it) {
        int chan = it * 2 + (tid >> 6);
        lds[pix * 264 + chan] = (f16)fb[(size_t)chan * HW_ + myidx];
    }
    __syncthreads();

    const int wave = tid >> 5;
    const int lane = tid & 31;
    const int half = lane >> 4;
    const int l16  = lane & 15;
    const int prow = wave * 16;

    v8f acc[16] = {};
    for (int cbi = 0; cbi < 8; ++cbi) {
        union { v16h h; unsigned u[8]; } a;
        const f16* arow = &lds[(prow + l16) * 264 + cbi * 32];
        #pragma unroll
        for (int v = 0; v < 8; ++v) {
            int kv = 16 * half + 2 * v;
            a.u[v] = *(const unsigned*)(arow + kv);
        }
        const f16* wf = WVF + (size_t)(cbi * 16) * 512 + lane * 16;
        #pragma unroll
        for (int t = 0; t < 16; ++t) {
            v16h bf = *(const v16h*)(wf + (size_t)t * 512);
            acc[t] = __builtin_amdgcn_wmma_f32_16x16x32_f16(false, a.h, false, bf, (short)0, acc[t], false, false);
        }
    }
    __syncthreads();   // done reading gather buffer; reuse LDS transposed

    #pragma unroll
    for (int t = 0; t < 16; ++t) {
        int   chan = t * 16 + l16;
        float bb   = bv[chan];
        #pragma unroll
        for (int r = 0; r < 8; ++r) {
            int key = prow + 8 * half + r;
            lds[chan * 72 + key] = (f16)(acc[t][r] + bb);
        }
    }
    __syncthreads();

    #pragma unroll
    for (int i = 0; i < 8; ++i) {
        int tile = wave * 8 + i;
        int kb2  = tile >> 4;
        int ct   = tile & 15;
        union { v16h h; unsigned u[8]; } o;
        const f16* col = &lds[(ct * 16 + l16) * 72 + kb2 * 32 + 16 * half];
        #pragma unroll
        for (int v = 0; v < 8; ++v) o.u[v] = *(const unsigned*)(col + 2 * v);
        int kb = (pbase >> 5) + kb2;
        *(v16h*)(outV + (size_t)(b * V_TILES + kb * 16 + ct) * 512 + lane * 16) = o.h;
    }
}

// ---------------------------------------------------------------------------
// Flash attention: per wave, 16 queries x all 4608 keys, online softmax.
// V fragment blocks (16KB/key-block) are TDM double-buffered into LDS and
// shared by all 4 waves; K fragments are small per-wave loads w/ prefetch.
// ---------------------------------------------------------------------------
__global__ __launch_bounds__(128)
void attn_kernel(const f16* __restrict__ Qf, const f16* __restrict__ Kf,
                 const f16* __restrict__ Vf, const int* __restrict__ idx_out,
                 const float* __restrict__ mask, const float* __restrict__ gamma,
                 float* __restrict__ out)
{
    __shared__ __align__(64) f16 ldsV[2][8192];   // 2 x 16KB V fragment buffers
    __shared__ f16 ldsP[4][16 * 34];              // per-wave P staging [q][k]

    const int tid  = threadIdx.x;
    const int wave = tid >> 5;
    const int lane = tid & 31;
    const int half = lane >> 4;
    const int l16  = lane & 15;

    const int b     = blockIdx.x / 72;
    const int qtile = (blockIdx.x % 72) * 4 + wave;
    const int qbase = qtile * 16;

    v16h qa = *(const v16h*)(Qf + (size_t)(b * QK_TILES + qtile) * 512 + lane * 16);

    v8f   acc[16] = {};
    float m[8], l[8];
    #pragma unroll
    for (int r = 0; r < 8; ++r) { m[r] = -1e30f; l[r] = 0.0f; }

    const f16* kbatch = Kf + (size_t)b * QK_TILES * 512;
    const f16* vbatch = Vf + (size_t)b * V_TILES * 512;

    const unsigned vOff0 = (unsigned)(uintptr_t)(void*)&ldsV[0][0];
    const unsigned vOff1 = (unsigned)(uintptr_t)(void*)&ldsV[1][0];

    // prologue: TDM-load V block 0
    if (wave == 0) tdm_load_1d(vbatch, vOff0, 16384);

    for (int kb = 0; kb < KBLOCKS; ++kb) {
        // barrier (end of previous iter) already guarantees buf[(kb+1)&1] free
        if (wave == 0) {
            if (kb + 1 < KBLOCKS) {
                tdm_load_1d(vbatch + (size_t)((kb + 1) * 16) * 512,
                            ((kb + 1) & 1) ? vOff1 : vOff0, 16384);
                __builtin_amdgcn_s_wait_tensorcnt(1);   // current block done
            } else {
                __builtin_amdgcn_s_wait_tensorcnt(0);
            }
        }
        __syncthreads();                                // current V block visible

        v16h k0 = *(const v16h*)(kbatch + (size_t)(2 * kb) * 512 + lane * 16);
        v16h k1 = *(const v16h*)(kbatch + (size_t)(2 * kb + 1) * 512 + lane * 16);
        if (kb + 1 < KBLOCKS)
            __builtin_prefetch(kbatch + (size_t)(2 * kb + 2) * 512, 0, 0);
        v8f s0 = {}; v8f s1 = {};
        s0 = __builtin_amdgcn_wmma_f32_16x16x32_f16(false, qa, false, k0, (short)0, s0, false, false);
        s1 = __builtin_amdgcn_wmma_f32_16x16x32_f16(false, qa, false, k1, (short)0, s1, false, false);

        // online softmax per query row (rows live in 16-lane half-groups)
        float sc8[8];
        #pragma unroll
        for (int r = 0; r < 8; ++r) {
            float mx = fmaxf(s0[r], s1[r]);
            mx = fmaxf(mx, __shfl_xor(mx, 1, 32));
            mx = fmaxf(mx, __shfl_xor(mx, 2, 32));
            mx = fmaxf(mx, __shfl_xor(mx, 4, 32));
            mx = fmaxf(mx, __shfl_xor(mx, 8, 32));
            float mn = fmaxf(m[r], mx);
            sc8[r] = __expf(m[r] - mn);
            m[r] = mn;
            float p0 = __expf(s0[r] - mn);
            float p1 = __expf(s1[r] - mn);
            float ps = p0 + p1;
            ps += __shfl_xor(ps, 1, 32);
            ps += __shfl_xor(ps, 2, 32);
            ps += __shfl_xor(ps, 4, 32);
            ps += __shfl_xor(ps, 8, 32);
            l[r] = l[r] * sc8[r] + ps;
            s0[r] = p0; s1[r] = p1;
        }
        // lazy O rescale: skip the 128 FMAs when no row max moved (wave-uniform)
        bool need = false;
        #pragma unroll
        for (int r = 0; r < 8; ++r) need |= (sc8[r] != 1.0f);
        if (__any(need)) {
            #pragma unroll
            for (int t = 0; t < 16; ++t)
                #pragma unroll
                for (int r = 0; r < 8; ++r) acc[t][r] *= sc8[r];
        }

        // P: D-layout -> A-layout via wave-private LDS
        f16* pw = &ldsP[wave][0];
        #pragma unroll
        for (int r = 0; r < 8; ++r) {
            int mrow = 8 * half + r;
            pw[mrow * 34 + l16]      = (f16)s0[r];
            pw[mrow * 34 + 16 + l16] = (f16)s1[r];
        }
        asm volatile("s_wait_dscnt 0x0" ::: "memory");
        union { v16h h; unsigned u[8]; } pa;
        const f16* pr = &ldsP[wave][l16 * 34];
        #pragma unroll
        for (int v = 0; v < 8; ++v) {
            int kv = 16 * half + 2 * v;
            pa.u[v] = *(const unsigned*)(pr + kv);
        }

        // O += P @ V^T from LDS (TDM-staged, shared by all 4 waves)
        const f16* vb = &ldsV[kb & 1][0] + lane * 16;
        #pragma unroll
        for (int t = 0; t < 16; ++t) {
            v16h vf = *(const v16h*)(vb + (size_t)t * 512);
            acc[t] = __builtin_amdgcn_wmma_f32_16x16x32_f16(false, pa.h, false, vf, (short)0, acc[t], false, false);
        }
        __syncthreads();   // all waves done with this V buffer before overwrite
    }

    // epilogue: r_v = O / l ; scatter into both output halves at idx_out
    const size_t outHalf = (size_t)B_ * C_ * HW_;
    float g = gamma[0];
    int   pg[8];
    float fac[8], rl[8];
    #pragma unroll
    for (int r = 0; r < 8; ++r) {
        int q  = qbase + 8 * half + r;
        int p  = idx_out[q];
        pg[r]  = p;
        float mk = mask[(size_t)b * HW_ + p];
        fac[r] = 1.0f + (1.0f - mk) * g;
        rl[r]  = 1.0f / l[r];
    }
    #pragma unroll
    for (int t = 0; t < 16; ++t) {
        int c = t * 16 + l16;
        float* base0 = out + (size_t)(b * C_ + c) * HW_;
        #pragma unroll
        for (int r = 0; r < 8; ++r) {
            float val = acc[t][r] * rl[r];
            base0[pg[r]] = val;
            base0[outHalf + pg[r]] = val * fac[r];
        }
    }
}

// ---------------------------------------------------------------------------
extern "C" void kernel_launch(void* const* d_in, const int* in_sizes, int n_in,
                              void* d_out, int out_size, void* d_ws, size_t ws_size,
                              hipStream_t stream)
{
    const float* f       = (const float*)d_in[0];
    const float* mask    = (const float*)d_in[1];
    const int*   idx_out = (const int*)d_in[2];
    const int*   idx_in  = (const int*)d_in[3];
    const float* Wq      = (const float*)d_in[4];
    const float* bq      = (const float*)d_in[5];
    const float* Wk      = (const float*)d_in[6];
    const float* bk      = (const float*)d_in[7];
    const float* Wv      = (const float*)d_in[8];
    const float* bv      = (const float*)d_in[9];
    const float* gamma   = (const float*)d_in[10];
    float* out = (float*)d_out;

    f16* Qf  = (f16*)((char*)d_ws + Q_OFF);
    f16* Kf  = (f16*)((char*)d_ws + K_OFF);
    f16* Vf  = (f16*)((char*)d_ws + V_OFF);
    f16* WQF = (f16*)((char*)d_ws + WQF_OFF);
    f16* WKF = (f16*)((char*)d_ws + WKF_OFF);
    f16* WVF = (f16*)((char*)d_ws + WVF_OFF);

    elw_kernel  <<<9216, 256, 0, stream>>>(f, mask, gamma, out);
    wfrag_kernel<<<4,  128, 0, stream>>>(Wq, WQF, 2);
    wfrag_kernel<<<4,  128, 0, stream>>>(Wk, WKF, 2);
    wfrag_kernel<<<32, 128, 0, stream>>>(Wv, WVF, 16);
    proj32_kernel<<<288, 128, 0, stream>>>(f, idx_out, WQF, bq, Qf, 0);
    proj32_kernel<<<288, 128, 0, stream>>>(f, idx_in,  WKF, bk, Kf, 1);
    projv_kernel <<<288, 128, 0, stream>>>(f, idx_in,  WVF, bv, Vf);
    attn_kernel  <<<288, 128, 0, stream>>>(Qf, Kf, Vf, idx_out, mask, gamma, out);
}
